// PolicyNet_53352083751181
// MI455X (gfx1250) — matrix operand order repaired and checked by reading, not compiled, not run on hardware
//
#include <hip/hip_runtime.h>

#define B_ 64
#define N_ 1024
#define H_ 64

typedef __attribute__((ext_vector_type(16))) __bf16        v16bf;
typedef __attribute__((ext_vector_type(8)))  float         v8f;
typedef __attribute__((ext_vector_type(4)))  float         f32x4;
typedef __attribute__((ext_vector_type(4)))  unsigned int  u32x4;

union BF16x16 {
  v16bf v;
  unsigned short s[16];
  unsigned int u[8];
  u32x4 q[2];
};

__device__ __forceinline__ unsigned short f32_to_bf16(float f) {
  unsigned u = __float_as_uint(f);
  u += 0x7FFFu + ((u >> 16) & 1u);   // round-to-nearest-even
  return (unsigned short)(u >> 16);
}
__device__ __forceinline__ float bf16_to_f32(unsigned short h) {
  return __uint_as_float(((unsigned)h) << 16);
}
// Truncating pack: two f32 -> dword of two bf16 (exact for 0/1 adjacency values).
// v_perm_b32: dst bytes = {a[3],a[2],b[3],b[2]}
__device__ __forceinline__ unsigned pack_bf16_trunc(float lo, float hi) {
  return __builtin_amdgcn_perm(__float_as_uint(hi), __float_as_uint(lo), 0x07060302u);
}

__device__ __forceinline__ v8f wmma_bf16(v16bf a, v16bf b, v8f c) {
  // (neg_a, A, neg_b, B, c_mod, C, reuse_a, reuse_b)
  return __builtin_amdgcn_wmma_f32_16x16x32_bf16(false, a, false, b, (short)0, c,
                                                 false, false);
}

// ---------------------------------------------------------------------------
// k0: prep bf16 transposed weights: W2T[c][r] = W2[r][c], Wm1bT[c][r] = Wm1[64+r][c]
// ---------------------------------------------------------------------------
__global__ void k0_prep(const float* __restrict__ W2, const float* __restrict__ Wm1,
                        unsigned short* __restrict__ W2T,
                        unsigned short* __restrict__ Wm1bT) {
  int tid = blockIdx.x * 256 + threadIdx.x;
  if (tid < 64 * 64) {
    int r = tid >> 6, c = tid & 63;
    W2T[c * 64 + r]   = f32_to_bf16(W2[r * 64 + c]);
    Wm1bT[c * 64 + r] = f32_to_bf16(Wm1[(64 + r) * 64 + c]);
  }
}

// ---------------------------------------------------------------------------
// k1: h1 = relu(X @ W1 + b1), stored transposed bf16 h1T[b][h][j]; detect curr.
// ---------------------------------------------------------------------------
__global__ __launch_bounds__(256) void k1_embed(const float* __restrict__ nf,
                                                const float* __restrict__ W1,
                                                const float* __restrict__ b1,
                                                unsigned short* __restrict__ h1T,
                                                int* __restrict__ curr) {
  const int blk = blockIdx.x;
  const int b = blk >> 2;
  const int j0 = (blk & 3) * 256;
  const int tid = threadIdx.x;
  const int j = j0 + tid;

  __shared__ __align__(16) unsigned short tile[64][256];  // [h][j-local]
  __shared__ float w1s[384];                              // 5*64 W1 + 64 b1

  if (tid < 384) w1s[tid] = (tid < 320) ? W1[tid] : b1[tid - 320];

  float x[5];
  const float* xp = nf + ((size_t)b * N_ + j) * 5;
#pragma unroll
  for (int f = 0; f < 5; f++) x[f] = xp[f];
  if (x[3] == 1.0f) curr[b] = j;  // exactly one node per batch has 1.0

  __syncthreads();

#pragma unroll 4
  for (int hc = 0; hc < 64; hc++) {
    float a = w1s[320 + hc];
#pragma unroll
    for (int f = 0; f < 5; f++) a += x[f] * w1s[f * 64 + hc];
    tile[hc][tid] = f32_to_bf16(fmaxf(a, 0.0f));
  }
  __syncthreads();

  // coalesced transposed writeout: thread -> (row hc, quarter), 128 B each
  const int hc = tid >> 2, part = tid & 3;
  const u32x4* s = (const u32x4*)(&tile[hc][part * 64]);
  u32x4* d = (u32x4*)(h1T + ((size_t)b * 64 + hc) * N_ + j0 + part * 64);
#pragma unroll
  for (int q = 0; q < 8; q++) d[q] = s[q];
}

// ---------------------------------------------------------------------------
// k2: h2 = relu((adj @ h1) @ W2 + b2)   -- the HBM-bound kernel.
// Grid: (N/128, B). 256 threads = 8 waves; wave w owns rows i0+16w..+15, all 64 cols.
// ---------------------------------------------------------------------------
__global__ __launch_bounds__(256) void k2_spmm(const float* __restrict__ adj,
                                               const unsigned short* __restrict__ h1T,
                                               const unsigned short* __restrict__ W2T,
                                               const float* __restrict__ b2,
                                               unsigned short* __restrict__ h2) {
  const int b = blockIdx.y;
  const int i0 = blockIdx.x * 128;
  const int tid = threadIdx.x;
  const int wave = tid >> 5;
  const int lane = tid & 31;
  const int m = lane & 15;   // A row / B-C-D column within 16
  const int g = lane >> 4;   // lane half -> K group
  const int iw = i0 + wave * 16;

  __shared__ __align__(16) unsigned short hT[64][264];    // h1T chunk: 64 h x 256 j (+8 pad)
  __shared__ __align__(16) unsigned short stg[8][16][72]; // per-wave 16x64 (+8 pad) relayout

  v8f acc[4];
#pragma unroll
  for (int nb = 0; nb < 4; nb++) {
    v8f z = {0.f, 0.f, 0.f, 0.f, 0.f, 0.f, 0.f, 0.f};
    acc[nb] = z;
  }

  const float* arow = adj + ((size_t)b * N_ + iw + m) * N_;

  for (int kc = 0; kc < N_; kc += 256) {
    // async global->LDS fill of h1T[:, kc:kc+256]: 2048 x 16B, 8 issues/thread,
    // tracked by ASYNCcnt (no VGPR staging round-trip).
#pragma unroll
    for (int r = 0; r < 8; r++) {
      int idx = r * 256 + tid;
      int hc = idx >> 5;  // 32 x 16B per 256-half row
      int o = idx & 31;
      unsigned lds_off = (unsigned)(unsigned long long)(void*)&hT[hc][o * 8];
      unsigned long long ga =
          (unsigned long long)(h1T + ((size_t)b * 64 + hc) * N_ + kc + o * 8);
      asm volatile("global_load_async_to_lds_b128 %0, %1, off"
                   :: "v"(lds_off), "v"(ga) : "memory");
    }
    asm volatile("s_wait_asynccnt 0x0" ::: "memory");
    __syncthreads();

    if (kc + 256 < N_) __builtin_prefetch(arow + kc + 256, 0, 0);  // global_prefetch_b8

#pragma unroll 2
    for (int k0 = kc; k0 < kc + 256; k0 += 32) {
      // A operand: adj fp32 -> bf16 (truncation, exact for 0/1), contiguous K = 16g + e order (same as B)
      const float* p1 = arow + k0 + g * 16;       // K = k0 + 16g + 0..7
      const float* p2 = arow + k0 + g * 16 + 8;   // K = k0 + 16g + 8..15
      f32x4 a0 = ((const f32x4*)p1)[0];
      f32x4 a1 = ((const f32x4*)p1)[1];
      f32x4 a2 = ((const f32x4*)p2)[0];
      f32x4 a3 = ((const f32x4*)p2)[1];
      BF16x16 A;
      A.u[0] = pack_bf16_trunc(a0.x, a0.y);
      A.u[1] = pack_bf16_trunc(a0.z, a0.w);
      A.u[2] = pack_bf16_trunc(a1.x, a1.y);
      A.u[3] = pack_bf16_trunc(a1.z, a1.w);
      A.u[4] = pack_bf16_trunc(a2.x, a2.y);
      A.u[5] = pack_bf16_trunc(a2.z, a2.w);
      A.u[6] = pack_bf16_trunc(a3.x, a3.y);
      A.u[7] = pack_bf16_trunc(a3.z, a3.w);

      const int kk = (k0 - kc) + g * 16;  // B: element e -> K = 16g + e
#pragma unroll
      for (int nb = 0; nb < 4; nb++) {
        BF16x16 Bv;
        const u32x4* bp = (const u32x4*)(&hT[nb * 16 + m][kk]);
        Bv.q[0] = bp[0];
        Bv.q[1] = bp[1];
        acc[nb] = wmma_bf16(A.v, Bv.v, acc[nb]);
      }
    }
    __syncthreads();
  }

  // --- fused h2 = relu(acc @ W2 + b2): re-layout C/D -> A via per-wave LDS ---
#pragma unroll
  for (int nb = 0; nb < 4; nb++)
#pragma unroll
    for (int r = 0; r < 8; r++)
      stg[wave][r + 8 * g][nb * 16 + m] = f32_to_bf16(acc[nb][r]);
  asm volatile("s_wait_dscnt 0x0" ::: "memory");

  v8f out[4];
#pragma unroll
  for (int nb = 0; nb < 4; nb++) {
    float bias = b2[nb * 16 + m];
    v8f c = {bias, bias, bias, bias, bias, bias, bias, bias};
    out[nb] = c;
  }
#pragma unroll
  for (int kt = 0; kt < 64; kt += 32) {
    BF16x16 A;
    A.q[0] = *(const u32x4*)(&stg[wave][m][kt + g * 16]);
    A.q[1] = *(const u32x4*)(&stg[wave][m][kt + g * 16 + 8]);
#pragma unroll
    for (int nb = 0; nb < 4; nb++) {
      BF16x16 Bv;
      const u32x4* wp = (const u32x4*)(W2T + (size_t)(nb * 16 + m) * 64 + kt + g * 16);
      Bv.q[0] = wp[0];
      Bv.q[1] = wp[1];
      out[nb] = wmma_bf16(A.v, Bv.v, out[nb]);
    }
  }

  // relu -> bf16 -> per-wave LDS -> coalesced 64B/lane global store
#pragma unroll
  for (int nb = 0; nb < 4; nb++)
#pragma unroll
    for (int r = 0; r < 8; r++)
      stg[wave][r + 8 * g][nb * 16 + m] = f32_to_bf16(fmaxf(out[nb][r], 0.0f));
  asm volatile("s_wait_dscnt 0x0" ::: "memory");

  {
    const int row = lane >> 1, part = lane & 1;
    unsigned short* dst = h2 + ((size_t)b * N_ + iw + row) * 64 + part * 32;
    const u32x4* s = (const u32x4*)(&stg[wave][row][part * 32]);
#pragma unroll
    for (int q = 0; q < 4; q++) ((u32x4*)dst)[q] = s[q];
  }
}

// ---------------------------------------------------------------------------
// k3a: t[b][h] = bm1[h] + h2[b, curr[b], :] @ Wm1[0:64].  64 threads per batch.
// ---------------------------------------------------------------------------
__global__ __launch_bounds__(64) void k3a_curr(const unsigned short* __restrict__ h2,
                                               const float* __restrict__ Wm1,
                                               const float* __restrict__ bm1,
                                               const int* __restrict__ curr,
                                               float* __restrict__ tvec) {
  const int b = blockIdx.x;
  const int t = threadIdx.x;
  __shared__ float ce[64];
  const int ci = curr[b];
  ce[t] = bf16_to_f32(h2[((size_t)b * N_ + ci) * 64 + t]);
  __syncthreads();
  float s = bm1[t];
#pragma unroll 8
  for (int k = 0; k < 64; k++) s += ce[k] * Wm1[k * 64 + t];
  tvec[b * 64 + t] = s;
}

// ---------------------------------------------------------------------------
// k3b: scores = relu(h2 @ Wm1b + t) @ Wm2 + bm2, masked by adj[b,curr,:].
// ---------------------------------------------------------------------------
__global__ __launch_bounds__(256) void k3b_score(const unsigned short* __restrict__ h2,
                                                 const unsigned short* __restrict__ Wm1bT,
                                                 const float* __restrict__ tvec,
                                                 const float* __restrict__ Wm2,
                                                 const float* __restrict__ bm2,
                                                 const float* __restrict__ adj,
                                                 const int* __restrict__ curr,
                                                 float* __restrict__ out) {
  const int b = blockIdx.y;
  const int i0 = blockIdx.x * 128;
  const int lane = threadIdx.x & 31;
  const int wave = threadIdx.x >> 5;
  const int m = lane & 15;
  const int g = lane >> 4;
  const int iw = i0 + wave * 16;

  const unsigned short* h2row = h2 + ((size_t)b * N_ + iw + m) * 64;

  v8f acc[4];
#pragma unroll
  for (int nb = 0; nb < 4; nb++) {
    float bias = tvec[b * 64 + nb * 16 + m];
    v8f c = {bias, bias, bias, bias, bias, bias, bias, bias};
    acc[nb] = c;
  }

#pragma unroll
  for (int kt = 0; kt < 64; kt += 32) {
    BF16x16 A;
    A.q[0] = *(const u32x4*)(h2row + kt + g * 16);
    A.q[1] = *(const u32x4*)(h2row + kt + g * 16 + 8);
#pragma unroll
    for (int nb = 0; nb < 4; nb++) {
      BF16x16 Bv;
      const u32x4* wp = (const u32x4*)(Wm1bT + (size_t)(nb * 16 + m) * 64 + kt + g * 16);
      Bv.q[0] = wp[0];
      Bv.q[1] = wp[1];
      acc[nb] = wmma_bf16(A.v, Bv.v, acc[nb]);
    }
  }

  // relu, scale by Wm2 column, row-reduce across the 16 lanes of each half
  float part[8];
#pragma unroll
  for (int r = 0; r < 8; r++) part[r] = 0.0f;
#pragma unroll
  for (int nb = 0; nb < 4; nb++) {
    float w = Wm2[nb * 16 + m];
#pragma unroll
    for (int r = 0; r < 8; r++) part[r] += fmaxf(acc[nb][r], 0.0f) * w;
  }
#pragma unroll
  for (int r = 0; r < 8; r++) {
    float p = part[r];
    p += __shfl_xor(p, 1, 32);
    p += __shfl_xor(p, 2, 32);
    p += __shfl_xor(p, 4, 32);
    p += __shfl_xor(p, 8, 32);
    part[r] = p;
  }

  if (m == 0) {  // lanes 0 (rows 0..7) and 16 (rows 8..15)
    const int ci = curr[b];
    const float* vrow = adj + ((size_t)b * N_ + ci) * N_;
    const float base = bm2[0];
#pragma unroll
    for (int r = 0; r < 8; r++) {
      int i = iw + 8 * g + r;
      float s = part[r] + base;
      out[b * N_ + i] = (vrow[i] == 0.0f) ? -1000000000.0f : s;
    }
  }
}

// ---------------------------------------------------------------------------
extern "C" void kernel_launch(void* const* d_in, const int* in_sizes, int n_in,
                              void* d_out, int out_size, void* d_ws, size_t ws_size,
                              hipStream_t stream) {
  const float* adj = (const float*)d_in[0];
  const float* nf  = (const float*)d_in[1];
  const float* W1  = (const float*)d_in[2];
  const float* b1  = (const float*)d_in[3];
  const float* W2  = (const float*)d_in[4];
  const float* b2  = (const float*)d_in[5];
  const float* Wm1 = (const float*)d_in[6];
  const float* bm1 = (const float*)d_in[7];
  const float* Wm2 = (const float*)d_in[8];
  const float* bm2 = (const float*)d_in[9];
  float* out = (float*)d_out;

  char* ws = (char*)d_ws;
  unsigned short* h1T   = (unsigned short*)(ws);              // 8 MB  bf16 [B][H][N]
  unsigned short* h2    = (unsigned short*)(ws + 8388608);    // 8 MB  bf16 [B][N][H]
  unsigned short* W2T   = (unsigned short*)(ws + 16777216);   // 8 KB
  unsigned short* Wm1bT = (unsigned short*)(ws + 16785408);   // 8 KB
  float*          tvec  = (float*)(ws + 16793600);            // 16 KB [B][H]
  int*            curr  = (int*)(ws + 16809984);              // 256 B [B]

  k0_prep<<<16, 256, 0, stream>>>(W2, Wm1, W2T, Wm1bT);
  k1_embed<<<B_ * 4, 256, 0, stream>>>(nf, W1, b1, h1T, curr);
  k2_spmm<<<dim3(N_ / 128, B_), 256, 0, stream>>>(adj, h1T, W2T, b2, h2);
  k3a_curr<<<B_, 64, 0, stream>>>(h2, Wm1, bm1, curr, tvec);
  k3b_score<<<dim3(N_ / 128, B_), 256, 0, stream>>>(h2, Wm1bT, tvec, Wm2, bm2, adj, curr, out);
}
